// TransformerBlock_43069932044586
// MI455X (gfx1250) — compile-verified, matrix operand order checked by reading, not run on hardware
//
#include <hip/hip_runtime.h>
#include <hip/hip_bf16.h>
#include <math.h>

typedef __attribute__((ext_vector_type(16))) _Float16 v16h;
typedef __attribute__((ext_vector_type(8)))  _Float16 v8h;
typedef __attribute__((ext_vector_type(8)))  float    v8f;

#define BB 2
#define SS 2048
#define DD 768
#define HH 12
#define HD 64
#define FF 3072
#define ROWS (BB*SS)          // 4096
#define EPS 1e-5f

#define BM 128
#define BN 128
#define BK 32

// ---------------------------------------------------------------------------
// Tensor Data Mover support (gfx1250). Falls back to vector loads if the
// builtin is unavailable (e.g. host pass of hipcc).
// ---------------------------------------------------------------------------
#ifndef __has_builtin
#define __has_builtin(x) 0
#endif
#if __has_builtin(__builtin_amdgcn_tensor_load_to_lds) && \
    __has_builtin(__builtin_amdgcn_s_wait_tensorcnt)
#define HAVE_TDM 1
#if defined(__has_include)
#if __has_include(<hip/amd_detail/amd_gfx1250_TDM.h>)
#define TDM_SIX_ARGS 1
#endif
#endif
#else
#define HAVE_TDM 0
#endif

#if HAVE_TDM
typedef unsigned int u32x4  __attribute__((ext_vector_type(4)));
typedef int          i32x4t __attribute__((ext_vector_type(4)));
typedef int          i32x8t __attribute__((ext_vector_type(8)));

// Issue a 2D TDM tile load: tileRows x tileCols elements (2 bytes each),
// global row stride = rowStrideElems, destination LDS byte offset lds_off.
static __device__ __forceinline__ void tdm_load_2d(const void* gsrc, unsigned lds_off,
                                                   int tileRows, int tileCols,
                                                   long long rowStrideElems) {
  unsigned long long ga = (unsigned long long)(size_t)gsrc;
  u32x4 g0;
  g0.x = 1u;                                                 // count = 1 valid D#
  g0.y = lds_off;                                            // lds_addr
  g0.z = (unsigned)ga;                                       // global_addr[31:0]
  g0.w = (unsigned)((ga >> 32) & 0x01ffffffu) | (2u << 30);  // global_addr[56:32], type=2
  unsigned long long str = (unsigned long long)rowStrideElems;
  unsigned td0 = 0x40000000u, td1 = 0x40000000u;             // huge extents: no OOB clip
  i32x8t g1;
  g1[0] = (int)(1u << 16);                                   // data_size = 2B
  g1[1] = (int)((td0 & 0xffffu) << 16);                      // tensor_dim0 lo
  g1[2] = (int)((td0 >> 16) | ((td1 & 0xffffu) << 16));      // tensor_dim0 hi | dim1 lo
  g1[3] = (int)((td1 >> 16) | ((unsigned)tileCols << 16));   // dim1 hi | tile_dim0
  g1[4] = (int)((unsigned)tileRows & 0xffffu);               // tile_dim1 (tile_dim2=0)
  g1[5] = (int)(unsigned)(str & 0xffffffffu);                // tensor_dim0_stride lo
  g1[6] = (int)(unsigned)((str >> 32) & 0xffffu);            // stride hi
  g1[7] = 0;
  i32x4t z = {};
#if defined(TDM_SIX_ARGS)
  i32x8t z8 = {};
  __builtin_amdgcn_tensor_load_to_lds(g0, g1, z, z, z8, 0);
#else
  __builtin_amdgcn_tensor_load_to_lds(g0, g1, z, z, 0);
#endif
}
#endif

static __device__ __forceinline__ v16h cat16(v8h lo, v8h hi) {
  return __builtin_shufflevector(lo, hi, 0,1,2,3,4,5,6,7,8,9,10,11,12,13,14,15);
}

// ---------------------------------------------------------------------------
// Weight conversion kernels: fp32 -> f16, producing TRANSPOSED (N-major)
// weights so the GEMM B-operand is contraction-contiguous.
// ---------------------------------------------------------------------------
__global__ __launch_bounds__(256) void cvt_whdk_T(const float* __restrict__ W,
                                                  _Float16* __restrict__ out) {
  // (H, D, HD) -> out[(h*HD+k)][D] : out[n*D + d] = W[h][d][k]
  int idx = blockIdx.x * 256 + threadIdx.x;
  if (idx >= DD * HH * HD) return;
  int n = idx / DD;
  int d = idx % DD;
  int h = n / HD, k = n % HD;
  out[idx] = (_Float16)W[((size_t)h * DD + d) * HD + k];
}

__global__ __launch_bounds__(256) void cvt_T(const float* __restrict__ W,
                                             _Float16* __restrict__ out,
                                             int R, int C) {
  // in [R][C] -> out [C][R]
  int idx = blockIdx.x * 256 + threadIdx.x;
  if (idx >= R * C) return;
  int c = idx / R, r = idx % R;
  out[idx] = (_Float16)W[(size_t)r * C + c];
}

// ---------------------------------------------------------------------------
// LayerNorm (row of 768) -> f16 output
// ---------------------------------------------------------------------------
__global__ __launch_bounds__(256) void ln_f16(const float* __restrict__ x,
                                              const float* __restrict__ g,
                                              const float* __restrict__ bta,
                                              _Float16* __restrict__ out) {
  __shared__ float red[256];
  int row = blockIdx.x, tid = threadIdx.x;
  const float* xr = x + (size_t)row * DD;
  float v0 = xr[tid], v1 = xr[tid + 256], v2 = xr[tid + 512];
  red[tid] = v0 + v1 + v2;
  __syncthreads();
  #pragma unroll
  for (int st = 128; st > 0; st >>= 1) {
    if (tid < st) red[tid] += red[tid + st];
    __syncthreads();
  }
  float mean = red[0] * (1.0f / DD);
  __syncthreads();
  float d0 = v0 - mean, d1 = v1 - mean, d2 = v2 - mean;
  red[tid] = d0 * d0 + d1 * d1 + d2 * d2;
  __syncthreads();
  #pragma unroll
  for (int st = 128; st > 0; st >>= 1) {
    if (tid < st) red[tid] += red[tid + st];
    __syncthreads();
  }
  float rs = rsqrtf(red[0] * (1.0f / DD) + EPS);
  _Float16* orow = out + (size_t)row * DD;
  orow[tid]       = (_Float16)(d0 * rs * g[tid]       + bta[tid]);
  orow[tid + 256] = (_Float16)(d1 * rs * g[tid + 256] + bta[tid + 256]);
  orow[tid + 512] = (_Float16)(d2 * rs * g[tid + 512] + bta[tid + 512]);
}

// ---------------------------------------------------------------------------
// WMMA GEMM: C[M,N] = act(A[M,K](f16) x BT[N,K](f16)^T + bias) (+ resid)
// 256 threads = 8 waves. Block tile 128x128, K-step 32.
// Wave grid 4(M) x 2(N); each wave: 2 A-frags x 4 B-frags -> 8 WMMA / K-step.
// B (weight) tiles are streamed by the Tensor Data Mover into double-buffered
// LDS; wave 0 issues the DMA and synchronizes with s_wait_tensorcnt.
// flags: bit0 = f16 output, bit1 = exact GELU
// ---------------------------------------------------------------------------
__global__ __launch_bounds__(256) void gemm_wmma(const _Float16* __restrict__ A,
                                                 const _Float16* __restrict__ BT,
                                                 const float* __restrict__ bias,
                                                 const float* __restrict__ resid,
                                                 void* __restrict__ Cout,
                                                 int M, int N, int K, int flags) {
  __shared__ __align__(16) _Float16 As[BM * BK];
#if HAVE_TDM
  __shared__ __align__(16) _Float16 Bsd[2][BN * BK];
#else
  __shared__ __align__(16) _Float16 Bs[BN * BK];
#endif
  int tid = threadIdx.x;
  int wid = tid >> 5, lane = tid & 31;
  int half = lane >> 4, r = lane & 15;
  int wm = wid & 3, wn = wid >> 2;
  int m0 = blockIdx.y * BM, n0 = blockIdx.x * BN;

  v8f acc[2][4];
  #pragma unroll
  for (int mi = 0; mi < 2; ++mi)
    #pragma unroll
    for (int ni = 0; ni < 4; ++ni) acc[mi][ni] = (v8f){};

  int ldRow = tid >> 1, ldCg = (tid & 1) * 16;   // 128 rows x 32 halves

#if HAVE_TDM
  if (wid == 0)
    tdm_load_2d(BT + (size_t)n0 * K, (unsigned)(size_t)&Bsd[0][0], BN, BK, K);
#endif

  for (int k0 = 0; k0 < K; k0 += BK) {
    const _Float16* ga = A + (size_t)(m0 + ldRow) * K + k0 + ldCg;
    v8h a0 = *(const v8h*)ga;
    v8h a1 = *(const v8h*)(ga + 8);
    if (k0 + BK < K) __builtin_prefetch(ga + BK, 0, 1);
    *(v8h*)&As[ldRow * BK + ldCg]     = a0;
    *(v8h*)&As[ldRow * BK + ldCg + 8] = a1;

#if HAVE_TDM
    int pb = (k0 / BK) & 1;
    if (wid == 0) {
      if (k0 + BK < K) {
        tdm_load_2d(BT + (size_t)n0 * K + (k0 + BK),
                    (unsigned)(size_t)&Bsd[pb ^ 1][0], BN, BK, K);
        __builtin_amdgcn_s_wait_tensorcnt(1);   // current tile has landed
      } else {
        __builtin_amdgcn_s_wait_tensorcnt(0);   // drain final tile
      }
    }
    const v8h* pB = (const v8h*)&Bsd[pb][0];
#else
    const _Float16* gb = BT + (size_t)(n0 + ldRow) * K + k0 + ldCg;
    v8h b0 = *(const v8h*)gb;
    v8h b1 = *(const v8h*)(gb + 8);
    if (k0 + BK < K) __builtin_prefetch(gb + BK, 0, 1);
    *(v8h*)&Bs[ldRow * BK + ldCg]     = b0;
    *(v8h*)&Bs[ldRow * BK + ldCg + 8] = b1;
    const v8h* pB = (const v8h*)Bs;
#endif
    __syncthreads();

    const v8h* pA = (const v8h*)As;
    v16h afr[2], bfr[4];
    #pragma unroll
    for (int mi = 0; mi < 2; ++mi) {
      int row = wm * 32 + mi * 16 + r;
      afr[mi] = cat16(pA[row * 4 + half], pA[row * 4 + 2 + half]);
    }
    #pragma unroll
    for (int ni = 0; ni < 4; ++ni) {
      int col = wn * 64 + ni * 16 + r;
      bfr[ni] = cat16(pB[col * 4 + half], pB[col * 4 + 2 + half]);
    }
    #pragma unroll
    for (int mi = 0; mi < 2; ++mi)
      #pragma unroll
      for (int ni = 0; ni < 4; ++ni)
        acc[mi][ni] = __builtin_amdgcn_wmma_f32_16x16x32_f16(
            false, afr[mi], false, bfr[ni], (short)0, acc[mi][ni], false, false);
    __syncthreads();
  }

  bool outf16 = flags & 1, doGelu = flags & 2;
  #pragma unroll
  for (int mi = 0; mi < 2; ++mi)
    #pragma unroll
    for (int ni = 0; ni < 4; ++ni)
      #pragma unroll
      for (int v = 0; v < 8; ++v) {
        int row = m0 + wm * 32 + mi * 16 + half * 8 + v;
        int col = n0 + wn * 64 + ni * 16 + r;
        float val = acc[mi][ni][v] + bias[col];
        if (doGelu) val = 0.5f * val * (1.0f + erff(val * 0.70710678f));
        if (resid) val += resid[(size_t)row * N + col];
        if (outf16) ((_Float16*)Cout)[(size_t)row * N + col] = (_Float16)val;
        else        ((float*)Cout)[(size_t)row * N + col] = val;
      }
}

// ---------------------------------------------------------------------------
// Causal flash attention. One block per (query-tile 64, head, batch).
// Q/K/V layout: [B, S, H*HD] f16. Output same layout, f16.
// Q,K,P row-major (contraction contiguous); V staged transposed.
// ---------------------------------------------------------------------------
__global__ __launch_bounds__(256) void attn_wmma(const _Float16* __restrict__ Qf,
                                                 const _Float16* __restrict__ Kf,
                                                 const _Float16* __restrict__ Vf,
                                                 _Float16* __restrict__ Of) {
  __shared__ __align__(16) _Float16 Qs[64 * 64];
  __shared__ __align__(16) _Float16 Ks[64 * 64];
  __shared__ __align__(16) _Float16 Ps[64 * 64];
  __shared__ __align__(16) _Float16 VsT[64 * 64];   // VsT[c][t]
  __shared__ float Sf[64][66];
  __shared__ float mrow[64], lrow[64], frow[64];

  int tid = threadIdx.x, wid = tid >> 5, lane = tid & 31;
  int half = lane >> 4, r = lane & 15;
  int wm = wid & 3, wn = wid >> 2;
  int iq = blockIdx.x, h = blockIdx.y, b = blockIdx.z;

  int ldRow = tid >> 2, ldCg = (tid & 3) * 16;  // 64 rows x 64 halves

  {
    size_t g = ((size_t)b * SS + (size_t)iq * 64 + ldRow) * DD + h * HD + ldCg;
    v8h q0 = *(const v8h*)(Qf + g);
    v8h q1 = *(const v8h*)(Qf + g + 8);
    *(v8h*)&Qs[ldRow * 64 + ldCg]     = q0;
    *(v8h*)&Qs[ldRow * 64 + ldCg + 8] = q1;
  }
  if (tid < 64) { mrow[tid] = -3.0e38f; lrow[tid] = 0.0f; }
  v8f o0 = {}, o1 = {};
  __syncthreads();

  int arow = wm * 16 + r, c0 = wn * 32 + r;
  const v8h* pQ = (const v8h*)Qs;
  const v8h* pK = (const v8h*)Ks;
  const v8h* pP = (const v8h*)Ps;
  const v8h* pV = (const v8h*)VsT;

  for (int j = 0; j <= iq; ++j) {
    {
      size_t g = ((size_t)b * SS + (size_t)j * 64 + ldRow) * DD + h * HD + ldCg;
      v8h k0v = *(const v8h*)(Kf + g);
      v8h k1v = *(const v8h*)(Kf + g + 8);
      v8h v0v = *(const v8h*)(Vf + g);
      v8h v1v = *(const v8h*)(Vf + g + 8);
      *(v8h*)&Ks[ldRow * 64 + ldCg]     = k0v;
      *(v8h*)&Ks[ldRow * 64 + ldCg + 8] = k1v;
      #pragma unroll
      for (int e = 0; e < 8; ++e) {
        VsT[(ldCg + e) * 64 + ldRow]     = v0v[e];
        VsT[(ldCg + 8 + e) * 64 + ldRow] = v1v[e];
      }
    }
    __syncthreads();

    // scores = Q . K^T
    v8f s0 = {}, s1 = {};
    #pragma unroll
    for (int kk = 0; kk < 2; ++kk) {
      v16h af  = cat16(pQ[arow * 8 + kk * 4 + half], pQ[arow * 8 + kk * 4 + 2 + half]);
      v16h bfA = cat16(pK[c0 * 8 + kk * 4 + half],   pK[c0 * 8 + kk * 4 + 2 + half]);
      v16h bfB = cat16(pK[(c0 + 16) * 8 + kk * 4 + half],
                       pK[(c0 + 16) * 8 + kk * 4 + 2 + half]);
      s0 = __builtin_amdgcn_wmma_f32_16x16x32_f16(false, af, false, bfA,
                                                  (short)0, s0, false, false);
      s1 = __builtin_amdgcn_wmma_f32_16x16x32_f16(false, af, false, bfB,
                                                  (short)0, s1, false, false);
    }
    #pragma unroll
    for (int v = 0; v < 8; ++v) {
      int lr = wm * 16 + half * 8 + v;
      int qg = iq * 64 + lr;
      float x0 = s0[v] * 0.125f, x1 = s1[v] * 0.125f;
      if (j * 64 + c0 > qg)      x0 = -1.0e30f;
      if (j * 64 + c0 + 16 > qg) x1 = -1.0e30f;
      Sf[lr][c0] = x0;
      Sf[lr][c0 + 16] = x1;
    }
    __syncthreads();

    // online softmax update (hw transcendental exp)
    if (tid < 64) {
      float mo = mrow[tid], tm = mo;
      #pragma unroll 8
      for (int c = 0; c < 64; ++c) tm = fmaxf(tm, Sf[tid][c]);
      float f = __expf(mo - tm), ts = 0.0f;
      #pragma unroll 8
      for (int c = 0; c < 64; ++c) {
        float p = __expf(Sf[tid][c] - tm);
        Ps[tid * 64 + c] = (_Float16)p;
        ts += p;
      }
      mrow[tid] = tm;
      frow[tid] = f;
      lrow[tid] = lrow[tid] * f + ts;
    }
    __syncthreads();

    #pragma unroll
    for (int v = 0; v < 8; ++v) {
      float f = frow[wm * 16 + half * 8 + v];
      o0[v] *= f; o1[v] *= f;
    }
    // O += P . V
    #pragma unroll
    for (int kk = 0; kk < 2; ++kk) {
      v16h af  = cat16(pP[arow * 8 + kk * 4 + half], pP[arow * 8 + kk * 4 + 2 + half]);
      v16h bfA = cat16(pV[c0 * 8 + kk * 4 + half],   pV[c0 * 8 + kk * 4 + 2 + half]);
      v16h bfB = cat16(pV[(c0 + 16) * 8 + kk * 4 + half],
                       pV[(c0 + 16) * 8 + kk * 4 + 2 + half]);
      o0 = __builtin_amdgcn_wmma_f32_16x16x32_f16(false, af, false, bfA,
                                                  (short)0, o0, false, false);
      o1 = __builtin_amdgcn_wmma_f32_16x16x32_f16(false, af, false, bfB,
                                                  (short)0, o1, false, false);
    }
    __syncthreads();
  }

  #pragma unroll
  for (int v = 0; v < 8; ++v) {
    int lr = wm * 16 + half * 8 + v;
    float inv = 1.0f / lrow[lr];
    size_t g = ((size_t)b * SS + (size_t)(iq * 64 + lr)) * DD + h * HD;
    Of[g + wn * 32 + r]      = (_Float16)(o0[v] * inv);
    Of[g + wn * 32 + 16 + r] = (_Float16)(o1[v] * inv);
  }
}

// ---------------------------------------------------------------------------
// Host-side launch
// ---------------------------------------------------------------------------
extern "C" void kernel_launch(void* const* d_in, const int* in_sizes, int n_in,
                              void* d_out, int out_size, void* d_ws, size_t ws_size,
                              hipStream_t stream) {
  const float* x     = (const float*)d_in[0];
  const float* Wq    = (const float*)d_in[1];
  const float* bq    = (const float*)d_in[2];
  const float* Wk    = (const float*)d_in[3];
  const float* bk    = (const float*)d_in[4];
  const float* Wv    = (const float*)d_in[5];
  const float* bv    = (const float*)d_in[6];
  const float* Wo    = (const float*)d_in[7];
  const float* bo    = (const float*)d_in[8];
  const float* ln1_g = (const float*)d_in[9];
  const float* ln1_b = (const float*)d_in[10];
  const float* ln2_g = (const float*)d_in[11];
  const float* ln2_b = (const float*)d_in[12];
  const float* W1    = (const float*)d_in[13];
  const float* b1    = (const float*)d_in[14];
  const float* W2    = (const float*)d_in[15];
  const float* b2    = (const float*)d_in[16];
  float* out = (float*)d_out;

  auto align = [](size_t v) { return (v + 255) & ~(size_t)255; };
  char* base = (char*)d_ws;
  size_t off = 0;
  auto slot = [&](size_t bytes) { char* p = base + off; off += align(bytes); return p; };
  _Float16* hf16  = (_Float16*)slot((size_t)ROWS * DD * 2);
  _Float16* WqT   = (_Float16*)slot((size_t)DD * DD * 2);   // [N=768][K=768]
  _Float16* WkT   = (_Float16*)slot((size_t)DD * DD * 2);
  _Float16* WvT   = (_Float16*)slot((size_t)DD * DD * 2);
  _Float16* WoT   = (_Float16*)slot((size_t)DD * DD * 2);
  _Float16* W1T   = (_Float16*)slot((size_t)DD * FF * 2);   // [N=3072][K=768]
  _Float16* W2T   = (_Float16*)slot((size_t)FF * DD * 2);   // [N=768][K=3072]
  _Float16* Qf    = (_Float16*)slot((size_t)ROWS * DD * 2);
  _Float16* Kf    = (_Float16*)slot((size_t)ROWS * DD * 2);
  _Float16* Vf    = (_Float16*)slot((size_t)ROWS * DD * 2);
  _Float16* At16  = (_Float16*)slot((size_t)ROWS * DD * 2);
  float*    X1    = (float*)   slot((size_t)ROWS * DD * 4);
  _Float16* h216  = (_Float16*)slot((size_t)ROWS * DD * 2);
  _Float16* H1    = (_Float16*)slot((size_t)ROWS * FF * 2);

  // 1) weight conversion (f32 -> f16, transposed to [N][K])
  cvt_whdk_T<<<(DD * HH * HD + 255) / 256, 256, 0, stream>>>(Wq, WqT);
  cvt_whdk_T<<<(DD * HH * HD + 255) / 256, 256, 0, stream>>>(Wk, WkT);
  cvt_whdk_T<<<(DD * HH * HD + 255) / 256, 256, 0, stream>>>(Wv, WvT);
  cvt_T<<<(DD * DD + 255) / 256, 256, 0, stream>>>(Wo, WoT, DD, DD);
  cvt_T<<<(DD * FF + 255) / 256, 256, 0, stream>>>(W1, W1T, DD, FF);
  cvt_T<<<(FF * DD + 255) / 256, 256, 0, stream>>>(W2, W2T, FF, DD);

  // 2) LN1
  ln_f16<<<ROWS, 256, 0, stream>>>(x, ln1_g, ln1_b, hf16);

  // 3) Q/K/V projections (f16 out, +bias)
  dim3 gQKV(DD / BN, ROWS / BM);
  gemm_wmma<<<gQKV, 256, 0, stream>>>(hf16, WqT, bq, nullptr, Qf, ROWS, DD, DD, 1);
  gemm_wmma<<<gQKV, 256, 0, stream>>>(hf16, WkT, bk, nullptr, Kf, ROWS, DD, DD, 1);
  gemm_wmma<<<gQKV, 256, 0, stream>>>(hf16, WvT, bv, nullptr, Vf, ROWS, DD, DD, 1);

  // 4) causal flash attention
  dim3 gAtt(SS / 64, HH, BB);
  attn_wmma<<<gAtt, 256, 0, stream>>>(Qf, Kf, Vf, At16);

  // 5) output projection + residual (f32 out): X1 = x + attn @ Wo + bo
  gemm_wmma<<<gQKV, 256, 0, stream>>>(At16, WoT, bo, x, X1, ROWS, DD, DD, 0);

  // 6) LN2
  ln_f16<<<ROWS, 256, 0, stream>>>(X1, ln2_g, ln2_b, h216);

  // 7) FFN up + GELU (f16 out)
  dim3 gF1(FF / BN, ROWS / BM);
  gemm_wmma<<<gF1, 256, 0, stream>>>(h216, W1T, b1, nullptr, H1, ROWS, FF, DD, 1 | 2);

  // 8) FFN down + bias + residual (f32 out to d_out)
  dim3 gF2(DD / BN, ROWS / BM);
  gemm_wmma<<<gF2, 256, 0, stream>>>(H1, W2T, b2, X1, out, ROWS, DD, FF, 0);

  (void)in_sizes; (void)n_in; (void)out_size; (void)ws_size;
}
